// PatchedModel_54485955117227
// MI455X (gfx1250) — compile-verified, matrix operand order checked by reading, not run on hardware
//
#include <hip/hip_runtime.h>
#include <hip/hip_bf16.h>
#include <stdint.h>

#define T_SEQ  2048
#define DMODEL 768
#define NHEAD  12
#define HDIM   64
#define WIN    64
#define NGLOB  16
#define NLOC   96     // union-local columns for a 16-query tile
#define NCOL   112    // valid key columns
#define NCOLP  128    // padded to multiple of 32 for WMMA K
#define KT_STEPS (DMODEL / 32)   // 24

typedef __attribute__((ext_vector_type(16))) __bf16 v16bf;
typedef __attribute__((ext_vector_type(8)))  __bf16 v8bf;
typedef __attribute__((ext_vector_type(8)))  float  v8f;

union AFrag { v16bf v; v8bf h[2]; };

__device__ __forceinline__ unsigned short f2bf(float f) {
    union { float f; unsigned int u; } v; v.f = f;
    unsigned int u = v.u;
    unsigned int r = u + 0x7FFFu + ((u >> 16) & 1u);
    return (unsigned short)(r >> 16);
}

__device__ __forceinline__ v8f wmma_bf16(v16bf a, v16bf b, v8f c) {
    return __builtin_amdgcn_wmma_f32_16x16x32_bf16(false, a, false, b, (short)0, c, false, false);
}

// k index held by (hi-group, half j) inside a 16x32 bf16 A fragment (ISA 7.12.2):
// lanes 0-15: halves 0-7 -> k 0-7, halves 8-15 -> k 16-23
// lanes 16-31: halves 0-7 -> k 8-15, halves 8-15 -> k 24-31
__device__ __forceinline__ int kmap(int hi, int j) { return (j < 8 ? j : j + 8) + hi * 8; }

// ---------------- weight packing: PB[(ntile*KT + kt)*512 + lane*16 + j] = W^T[k][n] = W[n][k]
__global__ void pack_weight_kernel(const float* __restrict__ W, unsigned short* __restrict__ PB) {
    int ntile = blockIdx.x;          // 0..47
    int kt    = blockIdx.y;          // 0..23
    size_t tilebase = ((size_t)ntile * KT_STEPS + kt) * 512;
    for (int e = threadIdx.x; e < 512; e += blockDim.x) {
        int lane = e >> 4, j = e & 15;
        int hi = lane >> 4, lo = lane & 15;
        int n = ntile * 16 + lo;
        int k = kt * 32 + kmap(hi, j);
        PB[tilebase + e] = f2bf(W[(size_t)n * DMODEL + k]);
    }
}

__global__ void conv_x_kernel(const float* __restrict__ X, unsigned short* __restrict__ Xbf, int n) {
    int i = blockIdx.x * blockDim.x + threadIdx.x;
    if (i < n) Xbf[i] = f2bf(X[i]);
}

// ---------------- gate scores: gs[t] = mask[t] ? h[t].Wg + bg : -1e9
__global__ void gate_kernel(const float* __restrict__ H, const float* __restrict__ Wg,
                            const float* __restrict__ bg, const unsigned char* __restrict__ mask,
                            float* __restrict__ gs) {
    int t = blockIdx.x;
    __shared__ float red[128];
    float s = 0.f;
    for (int i = threadIdx.x; i < DMODEL; i += 128) s += H[(size_t)t * DMODEL + i] * Wg[i];
    red[threadIdx.x] = s; __syncthreads();
    for (int st = 64; st > 0; st >>= 1) {
        if (threadIdx.x < st) red[threadIdx.x] += red[threadIdx.x + st];
        __syncthreads();
    }
    if (threadIdx.x == 0) gs[t] = mask[t] ? (red[0] + bg[0]) : -1e9f;
}

// ---------------- top-16 (iterative argmax, tie -> lower index, matches lax.top_k)
__global__ void topk_kernel(const float* __restrict__ gs, int* __restrict__ gidx) {
    __shared__ float sc[T_SEQ];
    __shared__ float rv[256];
    __shared__ int   ri[256];
    for (int i = threadIdx.x; i < T_SEQ; i += 256) sc[i] = gs[i];
    __syncthreads();
    for (int it = 0; it < NGLOB; ++it) {
        float bv = -3e38f; int bi = T_SEQ;
        for (int i = threadIdx.x; i < T_SEQ; i += 256) {
            float v = sc[i];
            if (v > bv || (v == bv && i < bi)) { bv = v; bi = i; }
        }
        rv[threadIdx.x] = bv; ri[threadIdx.x] = bi;
        __syncthreads();
        for (int st = 128; st > 0; st >>= 1) {
            if (threadIdx.x < st) {
                float ov = rv[threadIdx.x + st]; int oi = ri[threadIdx.x + st];
                if (ov > rv[threadIdx.x] || (ov == rv[threadIdx.x] && oi < ri[threadIdx.x])) {
                    rv[threadIdx.x] = ov; ri[threadIdx.x] = oi;
                }
            }
            __syncthreads();
        }
        if (threadIdx.x == 0) { gidx[it] = ri[0]; sc[ri[0]] = -3e38f; }
        __syncthreads();
    }
}

// ---------------- bf16 WMMA GEMM: Y[T,D] = X[T,D] @ W^T + bias, optional scale.
// Barrier-free register blocking: one wave owns a 32x32 output tile
// (2 row-fragments x 2 col-fragments -> 4 accumulators, 4 WMMA per K-step).
__global__ void gemm_wmma_kernel(const unsigned short* __restrict__ X,
                                 const unsigned short* __restrict__ PB,
                                 const float* __restrict__ bias,
                                 unsigned short* __restrict__ obf,
                                 float* __restrict__ of32, float scale) {
    int wave = threadIdx.x >> 5;
    int lane = threadIdx.x & 31;
    int hi = lane >> 4, lo = lane & 15;
    int t0  = blockIdx.x * 32;                 // 32 rows per wave
    int nt2 = blockIdx.y * 4 + wave;           // pair-of-16-col tiles: 0..23
    int ntile0 = nt2 * 2;                      // 0..46 (even)

    const unsigned short* arow0 = X + (size_t)(t0 + lo) * DMODEL;
    const unsigned short* arow1 = X + (size_t)(t0 + 16 + lo) * DMODEL;
    const unsigned short* pb0 = PB + ((size_t)ntile0 * KT_STEPS * 32 + lane) * 16;
    const unsigned short* pb1 = pb0 + (size_t)KT_STEPS * 512;   // next 16-col tile

    v8f z = {0.f,0.f,0.f,0.f,0.f,0.f,0.f,0.f};
    v8f c00 = z, c01 = z, c10 = z, c11 = z;

    for (int kt = 0; kt < KT_STEPS; ++kt) {
        AFrag a0, a1;
        a0.h[0] = *(const v8bf*)(arow0 + kt * 32 + hi * 8);
        a0.h[1] = *(const v8bf*)(arow0 + kt * 32 + 16 + hi * 8);
        a1.h[0] = *(const v8bf*)(arow1 + kt * 32 + hi * 8);
        a1.h[1] = *(const v8bf*)(arow1 + kt * 32 + 16 + hi * 8);
        v16bf b0 = *(const v16bf*)(pb0 + (size_t)kt * 512);
        v16bf b1 = *(const v16bf*)(pb1 + (size_t)kt * 512);
        if (kt + 4 < KT_STEPS) {   // cover first-touch latency on the B stream
            __builtin_prefetch(pb0 + (size_t)(kt + 4) * 512, 0, 1);
            __builtin_prefetch(pb1 + (size_t)(kt + 4) * 512, 0, 1);
        }
        c00 = wmma_bf16(a0.v, b0, c00);
        c01 = wmma_bf16(a0.v, b1, c01);
        c10 = wmma_bf16(a1.v, b0, c10);
        c11 = wmma_bf16(a1.v, b1, c11);
    }

    v8f accs[4] = {c00, c01, c10, c11};
    for (int q = 0; q < 4; ++q) {
        int rbase = t0 + (q >> 1) * 16 + hi * 8;
        int n = (ntile0 + (q & 1)) * 16 + lo;
        float bs = bias[n];
        for (int r = 0; r < 8; ++r) {
            float val = (accs[q][r] + bs) * scale;
            int m = rbase + r;
            if (obf)  obf[(size_t)m * DMODEL + n]  = f2bf(val);
            if (of32) of32[(size_t)m * DMODEL + n] = val;
        }
    }
}

// ---------------- sparse attention: one wave per (16-query tile, head)
__global__ void attn_kernel(const unsigned short* __restrict__ Qbf,
                            const unsigned short* __restrict__ Kbf,
                            const unsigned short* __restrict__ Vbf,
                            const int* __restrict__ gidx,
                            const unsigned char* __restrict__ mask,
                            unsigned short* __restrict__ obf) {
    int t0 = blockIdx.x * 16;
    int h  = blockIdx.y;
    int lane = threadIdx.x;
    int hi = lane >> 4, lo = lane & 15;

    int s0 = t0 - 32;
    if (s0 < 0) s0 = 0;
    if (s0 > T_SEQ - WIN) s0 = T_SEQ - WIN;

    __shared__ int keyidx[NCOLP];
    __shared__ unsigned char okcol[NCOLP];
    __shared__ __align__(16) unsigned short P[16 * NCOLP];

    for (int i = lane; i < NCOLP; i += 32) {
        int key; unsigned char ok;
        if (i < NGLOB)      { key = gidx[i]; ok = mask[key]; }
        else if (i < NCOL)  { int u = i - NGLOB; key = s0 + u;
                              if (key >= T_SEQ) { key = T_SEQ - 1; ok = 0; } else ok = mask[key]; }
        else                { key = 0; ok = 0; }
        keyidx[i] = key; okcol[i] = ok;
    }
    __syncthreads();

    size_t hoff = (size_t)h * HDIM;

    // Q tile 16x64 as two 16x32 A fragments
    AFrag qa0, qa1;
    {
        const unsigned short* qrow = Qbf + (size_t)(t0 + lo) * DMODEL + hoff;
        qa0.h[0] = *(const v8bf*)(qrow + hi * 8);
        qa0.h[1] = *(const v8bf*)(qrow + 16 + hi * 8);
        qa1.h[0] = *(const v8bf*)(qrow + 32 + hi * 8);
        qa1.h[1] = *(const v8bf*)(qrow + 48 + hi * 8);
    }

    // scores: 16 x 112 (7 N-tiles)
    v8f sc[7];
    for (int nt = 0; nt < 7; ++nt) {
        int key = keyidx[nt * 16 + lo];
        const unsigned short* krow = Kbf + (size_t)key * DMODEL + hoff;
        AFrag b0, b1;
        b0.h[0] = *(const v8bf*)(krow + hi * 8);
        b0.h[1] = *(const v8bf*)(krow + 16 + hi * 8);
        b1.h[0] = *(const v8bf*)(krow + 32 + hi * 8);
        b1.h[1] = *(const v8bf*)(krow + 48 + hi * 8);
        v8f c = {0.f,0.f,0.f,0.f,0.f,0.f,0.f,0.f};
        c = wmma_bf16(qa0.v, b0.v, c);
        c = wmma_bf16(qa1.v, b1.v, c);
        sc[nt] = c;
    }

    // per-row masking + softmax (rows for this lane: m = hi*8 + r; row group = 16 lanes)
    for (int r = 0; r < 8; ++r) {
        int t = t0 + hi * 8 + r;
        int st = t - 32;
        if (st < 0) st = 0;
        if (st > T_SEQ - WIN) st = T_SEQ - WIN;
        int strel = st - s0;
        for (int nt = 0; nt < 7; ++nt) {
            int cidx = nt * 16 + lo;
            bool ok = okcol[cidx] != 0;
            if (cidx >= NGLOB) {
                int u = cidx - NGLOB;
                ok = ok && (u >= strel) && (u < strel + WIN);
            }
            float s = ok ? sc[nt][r] : -1e9f;
            sc[nt][r] = s;
        }
        float mx = -3e38f;
        for (int nt = 0; nt < 7; ++nt) mx = fmaxf(mx, sc[nt][r]);
        for (int off = 1; off < 16; off <<= 1) mx = fmaxf(mx, __shfl_xor(mx, off, 32));
        float sum = 0.f;
        for (int nt = 0; nt < 7; ++nt) {
            float p = __expf(sc[nt][r] - mx);
            sc[nt][r] = p; sum += p;
        }
        for (int off = 1; off < 16; off <<= 1) sum += __shfl_xor(sum, off, 32);
        float inv = 1.0f / sum;
        for (int nt = 0; nt < 7; ++nt) sc[nt][r] *= inv;
    }

    // bounce probs through LDS to re-layout C -> A (16 x 128, padded K zeroed)
    for (int i = lane; i < (16 * NCOLP) / 2; i += 32) ((unsigned int*)P)[i] = 0u;
    __syncthreads();
    for (int r = 0; r < 8; ++r) {
        int m = hi * 8 + r;
        for (int nt = 0; nt < 7; ++nt) P[m * NCOLP + nt * 16 + lo] = f2bf(sc[nt][r]);
    }
    __syncthreads();

    // out = probs(16x128) @ V_sel(128x64)
    v8f z = {0.f,0.f,0.f,0.f,0.f,0.f,0.f,0.f};
    v8f acc[4] = {z, z, z, z};
    for (int kc = 0; kc < 4; ++kc) {
        AFrag ap;
        ap.h[0] = *(const v8bf*)&P[lo * NCOLP + kc * 32 + hi * 8];
        ap.h[1] = *(const v8bf*)&P[lo * NCOLP + kc * 32 + 16 + hi * 8];
        for (int nd = 0; nd < 4; ++nd) {
            union { v16bf v; unsigned short u[16]; } bv;
            int n = nd * 16 + lo;
            for (int j = 0; j < 16; ++j) {
                int k = kc * 32 + kmap(hi, j);
                int key = keyidx[k];
                bv.u[j] = Vbf[(size_t)key * DMODEL + hoff + n];
            }
            acc[nd] = wmma_bf16(ap.v, bv.v, acc[nd]);
        }
    }

    for (int nd = 0; nd < 4; ++nd) {
        int n = nd * 16 + lo;
        for (int r = 0; r < 8; ++r) {
            int m = t0 + hi * 8 + r;
            obf[(size_t)m * DMODEL + hoff + n] = f2bf(acc[nd][r]);
        }
    }
}

extern "C" void kernel_launch(void* const* d_in, const int* in_sizes, int n_in,
                              void* d_out, int out_size, void* d_ws, size_t ws_size,
                              hipStream_t stream) {
    const float* hidden = (const float*)d_in[0];
    const float* Wq = (const float*)d_in[1];
    const float* bq = (const float*)d_in[2];
    const float* Wk = (const float*)d_in[3];
    const float* bk = (const float*)d_in[4];
    const float* Wv = (const float*)d_in[5];
    const float* bv = (const float*)d_in[6];
    const float* Wo = (const float*)d_in[7];
    const float* bo = (const float*)d_in[8];
    const float* Wg = (const float*)d_in[9];
    const float* bg = (const float*)d_in[10];
    const unsigned char* mask = (const unsigned char*)d_in[11];

    char* base = (char*)d_ws;
    auto nxt = [&](size_t n) { char* p = base; base += (n + 255) & ~(size_t)255; return p; };
    const size_t TD2 = (size_t)T_SEQ * DMODEL * 2;
    const size_t DD2 = (size_t)DMODEL * DMODEL * 2;
    unsigned short* Xbf    = (unsigned short*)nxt(TD2);
    unsigned short* PWq    = (unsigned short*)nxt(DD2);
    unsigned short* PWk    = (unsigned short*)nxt(DD2);
    unsigned short* PWv    = (unsigned short*)nxt(DD2);
    unsigned short* PWo    = (unsigned short*)nxt(DD2);
    unsigned short* Qbf    = (unsigned short*)nxt(TD2);
    unsigned short* Kbf    = (unsigned short*)nxt(TD2);
    unsigned short* Vbf    = (unsigned short*)nxt(TD2);
    unsigned short* AttObf = (unsigned short*)nxt(TD2);
    float* gsc = (float*)nxt((size_t)T_SEQ * 4);
    int*   gix = (int*)nxt(256);

    dim3 packGrid(DMODEL / 16, DMODEL / 32);
    pack_weight_kernel<<<packGrid, 256, 0, stream>>>(Wq, PWq);
    pack_weight_kernel<<<packGrid, 256, 0, stream>>>(Wk, PWk);
    pack_weight_kernel<<<packGrid, 256, 0, stream>>>(Wv, PWv);
    pack_weight_kernel<<<packGrid, 256, 0, stream>>>(Wo, PWo);

    int nelem = T_SEQ * DMODEL;
    conv_x_kernel<<<(nelem + 255) / 256, 256, 0, stream>>>(hidden, Xbf, nelem);

    gate_kernel<<<T_SEQ, 128, 0, stream>>>(hidden, Wg, bg, mask, gsc);
    topk_kernel<<<1, 256, 0, stream>>>(gsc, gix);

    dim3 gemmGrid(T_SEQ / 32, DMODEL / 128);   // (64, 6), 4 waves/block
    gemm_wmma_kernel<<<gemmGrid, 128, 0, stream>>>(Xbf, PWq, bq, Qbf, nullptr, 0.125f); // d^-1/2
    gemm_wmma_kernel<<<gemmGrid, 128, 0, stream>>>(Xbf, PWk, bk, Kbf, nullptr, 1.0f);
    gemm_wmma_kernel<<<gemmGrid, 128, 0, stream>>>(Xbf, PWv, bv, Vbf, nullptr, 1.0f);

    attn_kernel<<<dim3(T_SEQ / 16, NHEAD), 32, 0, stream>>>(Qbf, Kbf, Vbf, gix, mask, AttObf);

    gemm_wmma_kernel<<<gemmGrid, 128, 0, stream>>>(AttObf, PWo, bo, nullptr, (float*)d_out, 1.0f);
}
